// RANSACMatcher_35218731827995
// MI455X (gfx1250) — compile-verified, matrix operand order checked by reading, not run on hardware
//
#include <hip/hip_runtime.h>

typedef float v2f __attribute__((ext_vector_type(2)));
typedef float v8f __attribute__((ext_vector_type(8)));
typedef _Float16 v16h __attribute__((ext_vector_type(16)));

constexpr int kB = 4, kN = 1024, kM = 1024, kD = 128;
constexpr int kIT = 16, kRN = 614, kRM = 614;
constexpr float kRatioThr = 0.6f;
constexpr float kMaxMatchedRatio = 0.01f;
constexpr float kRansacThr = 0.75f;
constexpr int kXS = 132;  // padded LDS row stride (floats): 132 % 64 = 4 banks/row,
                          // 132*4 = 528 B = 33*16 B keeps B128 async chunks aligned

#if defined(__has_builtin)
#if __has_builtin(__builtin_amdgcn_wmma_f32_16x16x4_f32)
#define HAVE_WMMA_F32 1
#endif
#endif

__device__ __forceinline__ float dinf() { return __builtin_inff(); }
__device__ __forceinline__ float dnan() { return __builtin_nanf(""); }

// ---------------------------------------------------------------- norms
// one wave per row of 128 floats: ||row||^2
__global__ void norms_kernel(const float* __restrict__ src,
                             float* __restrict__ dst, int rows) {
  int gid = blockIdx.x * blockDim.x + threadIdx.x;
  int wave = gid >> 5, lane = gid & 31;
  if (wave >= rows) return;
  const float* r = src + (size_t)wave * kD;
  float s = 0.f;
  for (int i = lane; i < kD; i += 32) { float v = r[i]; s += v * v; }
  for (int o = 16; o > 0; o >>= 1) s += __shfl_down(s, o, 32);
  if (lane == 0) dst[wave] = s;
}

// ---------------------------------------------------------------- WMMA GEMM + dist
// Block tile 32x64, K=128 fully staged in LDS via async global->LDS (ASYNCcnt),
// then 8 waves compute 16x16 WMMA tiles fed by ds_load. Fused epilogue:
// dist = sqrt(max(|x|^2+|y|^2-2P, 1e-12)), inf where mask<=0.
__global__ __launch_bounds__(256) void gemm_dist_kernel(
    const float* __restrict__ xd, const float* __restrict__ yd,
    const float* __restrict__ mask, const float* __restrict__ xn,
    const float* __restrict__ yn, float* __restrict__ dist) {
  __shared__ __align__(16) float sx[32 * kXS];
  __shared__ __align__(16) float sy[64 * kXS];
  const int b = blockIdx.z;
  const int n0blk = blockIdx.y * 32;
  const int m0blk = blockIdx.x * 64;
  const int tid = threadIdx.x;

  // ---- async stage: x tile (32x128) + y tile (64x128), 16B chunks
  {
    const char* gx = (const char*)(xd + ((size_t)b * kN + n0blk) * kD);
    unsigned lx = (unsigned)(uintptr_t)sx;  // flat addr[31:0] == LDS address
    for (int c = tid; c < 32 * 32; c += 256) {
      int row = c >> 5, col = c & 31;
      const char* g = gx + (size_t)row * (kD * 4) + col * 16;
      unsigned l = lx + row * (kXS * 4) + col * 16;
      asm volatile("global_load_async_to_lds_b128 %0, %1, off"
                   :: "v"(l), "v"(g) : "memory");
    }
    const char* gy = (const char*)(yd + ((size_t)b * kM + m0blk) * kD);
    unsigned ly = (unsigned)(uintptr_t)sy;
    for (int c = tid; c < 64 * 32; c += 256) {
      int row = c >> 5, col = c & 31;
      const char* g = gy + (size_t)row * (kD * 4) + col * 16;
      unsigned l = ly + row * (kXS * 4) + col * 16;
      asm volatile("global_load_async_to_lds_b128 %0, %1, off"
                   :: "v"(l), "v"(g) : "memory");
    }
    asm volatile("s_wait_asynccnt 0x0" ::: "memory");
  }
  __syncthreads();

  const int lane = tid & 31;
  const int wave = tid >> 5;
  const int wr = (wave >> 2) * 16;  // row offset within 32-row tile
  const int wc = (wave & 3) * 16;   // col offset within 64-col tile
  const int l = lane & 15, half = lane >> 4;
  const float* xrow = sx + (size_t)(wr + l) * kXS;
  const float* yrow = sy + (size_t)(wc + l) * kXS;
  v8f acc = {};
#if HAVE_WMMA_F32
  // A(16x4): lane<16 -> M=lane, VGPR0/1 = K=0/1 ; lane>=16 -> K=2/3
  // B(4x16): lane = N index, same half/K-pair mapping (symmetric)
  for (int k = 0; k < kD; k += 4) {
    v2f a = *(const v2f*)(xrow + k + 2 * half);
    v2f bb = *(const v2f*)(yrow + k + 2 * half);
    acc = __builtin_amdgcn_wmma_f32_16x16x4_f32(
        false, a, false, bb, (short)0, acc, false, false);
  }
#else
  for (int k0 = 0; k0 < kD; k0 += 32) {
    v16h a, bb;
    for (int v = 0; v < 8; ++v) {
      int kb = (v < 4 ? 2 * v : 16 + 2 * (v - 4)) + half * 8;
      a[2 * v] = (_Float16)xrow[k0 + kb];
      a[2 * v + 1] = (_Float16)xrow[k0 + kb + 1];
      bb[2 * v] = (_Float16)yrow[k0 + kb];
      bb[2 * v + 1] = (_Float16)yrow[k0 + kb + 1];
    }
    acc = __builtin_amdgcn_wmma_f32_16x16x32_f16(
        false, a, false, bb, (short)0, acc, false, false);
  }
#endif
  // C/D layout: VGPR r -> M = r (lanes 0-15), M = r+8 (lanes 16-31); N = lane&15
  for (int r = 0; r < 8; ++r) {
    int n = n0blk + wr + r + 8 * half;
    int m = m0blk + wc + l;
    size_t idx = ((size_t)b * kN + n) * kM + m;
    float d2 = xn[b * kN + n] + yn[b * kM + m] - 2.0f * acc[r];
    float d = sqrtf(fmaxf(d2, 1e-12f));
    if (!(mask[idx] > 0.f)) d = dinf();
    dist[idx] = d;
  }
}

// ---------------------------------------------------------------- per-batch 1% quantile
// rank-selection by value bisection; dist stays L2-resident (16MB << 192MB L2)
__global__ __launch_bounds__(1024) void quant_kernel(
    const float* __restrict__ dist, float* __restrict__ thrq) {
  const int b = blockIdx.x, tid = threadIdx.x;
  const size_t base = (size_t)b * kN * kM;
  const int NM = kN * kM;
  __shared__ float rf[1024];
  __shared__ int ri[1024];
  float mx = 0.f;
  for (int i = tid; i < NM; i += 1024) {
    float v = dist[base + i];
    if (v < 3.0e38f && v > mx) mx = v;
  }
  rf[tid] = mx; __syncthreads();
  for (int o = 512; o > 0; o >>= 1) {
    if (tid < o) rf[tid] = fmaxf(rf[tid], rf[tid + o]);
    __syncthreads();
  }
  mx = rf[0]; __syncthreads();
  const double h = (double)kMaxMatchedRatio * (NM - 1);
  const int k = (int)h;
  const float frac = (float)(h - (double)k);
  float lo = 0.f, hi = fmaxf(mx, 1e-6f) * 1.000001f;
  for (int itb = 0; itb < 32; ++itb) {
    float mid = 0.5f * (lo + hi);
    int c = 0;
    for (int i = tid; i < NM; i += 1024) c += (dist[base + i] <= mid) ? 1 : 0;
    ri[tid] = c; __syncthreads();
    for (int o = 512; o > 0; o >>= 1) {
      if (tid < o) ri[tid] += ri[tid + o];
      __syncthreads();
    }
    c = ri[0]; __syncthreads();
    if (c >= k + 1) hi = mid; else lo = mid;
  }
  float vk = hi;
  int c = 0; float nxt = dinf();
  for (int i = tid; i < NM; i += 1024) {
    float v = dist[base + i];
    c += (v <= vk) ? 1 : 0;
    if (v > vk && v < nxt) nxt = v;
  }
  ri[tid] = c; rf[tid] = nxt; __syncthreads();
  for (int o = 512; o > 0; o >>= 1) {
    if (tid < o) { ri[tid] += ri[tid + o]; rf[tid] = fminf(rf[tid], rf[tid + o]); }
    __syncthreads();
  }
  if (tid == 0) {
    float vk1 = (ri[0] >= k + 2) ? vk : rf[0];
    thrq[b] = vk + frac * (vk1 - vk);
  }
}

// ---------------------------------------------------------------- row / col stats
__device__ __forceinline__ void top2_reduce(float& m1, float& m2, int& i1) {
  for (int o = 16; o > 0; o >>= 1) {
    float om1 = __shfl_down(m1, o, 32);
    float om2 = __shfl_down(m2, o, 32);
    int oi1 = __shfl_down(i1, o, 32);
    if (om1 < m1 || (om1 == m1 && oi1 < i1)) {
      m2 = fminf(m1, om2); m1 = om1; i1 = oi1;
    } else {
      m2 = fminf(m2, om1);
    }
  }
}

__global__ void rowstat_kernel(const float* __restrict__ dist,
                               int* __restrict__ idx, int* __restrict__ rt) {
  int gid = blockIdx.x * blockDim.x + threadIdx.x;
  int row = gid >> 5, lane = gid & 31;
  if (row >= kB * kN) return;
  const float* r = dist + (size_t)row * kM;
  float m1 = dinf(), m2 = dinf(); int i1 = 0x7fffffff;
  for (int m = lane; m < kM; m += 32) {
    float v = r[m];
    if (v < m1) { m2 = m1; m1 = v; i1 = m; }
    else if (v < m2) m2 = v;
  }
  top2_reduce(m1, m2, i1);
  if (lane == 0) {
    idx[row] = i1;
    rt[row] = (m1 / (1e-6f + m2) < kRatioThr) ? 1 : 0;
  }
}

__global__ void colstat_kernel(const float* __restrict__ dist,
                               int* __restrict__ idx, int* __restrict__ rt) {
  int gid = blockIdx.x * blockDim.x + threadIdx.x;
  int col = gid >> 5, lane = gid & 31;
  if (col >= kB * kM) return;
  int b = col / kM, m = col % kM;
  const float* base = dist + (size_t)b * kN * kM + m;
  float m1 = dinf(), m2 = dinf(); int i1 = 0x7fffffff;
  for (int n = lane; n < kN; n += 32) {
    float v = base[(size_t)n * kM];
    if (v < m1) { m2 = m1; m1 = v; i1 = n; }
    else if (v < m2) m2 = v;
  }
  top2_reduce(m1, m2, i1);
  if (lane == 0) {
    idx[col] = i1;
    rt[col] = (m1 / (1e-6f + m2) < kRatioThr) ? 1 : 0;
  }
}

// ---------------------------------------------------------------- mutual-NN mask
__global__ void combine_kernel(const float* __restrict__ dist,
                               const float* __restrict__ mask,
                               const int* __restrict__ ridx, const int* __restrict__ rrt,
                               const int* __restrict__ cidx, const int* __restrict__ crt,
                               const float* __restrict__ thrq,
                               unsigned char* __restrict__ w) {
  int row = blockIdx.x;
  int b = row / kN, n = row % kN;
  float thr = thrq[b];
  int ri = ridx[row], rr = rrt[row];
  size_t base = (size_t)row * kM;
  for (int m = threadIdx.x; m < kM; m += blockDim.x) {
    float dv = dist[base + m];
    bool wv = rr && (ri == m) && crt[b * kM + m] && (cidx[b * kM + m] == n) &&
              (mask[base + m] > 0.f) && (dv <= thr);
    w[base + m] = wv ? 1 : 0;
  }
}

// ---------------------------------------------------------------- permutations
__global__ void perm_kernel(int* __restrict__ p1, int* __restrict__ p2) {
  int t = threadIdx.x;
  if (t >= 2 * kIT) return;
  int* p = (t < kIT) ? (p1 + t * kRN) : (p2 + (t - kIT) * kRM);
  unsigned s = 0x9E3779B9u * (unsigned)(t + 1) + 12345u;
  for (int i = 0; i < kRN; ++i) p[i] = i;
  for (int i = kRN - 1; i > 0; --i) {
    s = s * 1664525u + 1013904223u;
    int j = (int)(s % (unsigned)(i + 1));
    int tmp = p[i]; p[i] = p[j]; p[j] = tmp;
  }
}

// ---------------------------------------------------------------- solver (3x3 model)
__global__ __launch_bounds__(256) void solver_kernel(
    const float* __restrict__ xk, const float* __restrict__ yk,
    const unsigned char* __restrict__ w, const int* __restrict__ p1,
    const int* __restrict__ p2, float* __restrict__ models) {
  int wg = blockIdx.x;
  int b = wg / kIT, it = wg % kIT;
  __shared__ int sp2[kRM];
  __shared__ float sy[kRM * 3];
  __shared__ float snum[9];
  __shared__ float sden;
  for (int m = threadIdx.x; m < kRM; m += blockDim.x) {
    int p = p2[it * kRM + m];
    sp2[m] = p;
    sy[3 * m + 0] = yk[((size_t)b * kM + p) * 3 + 0];
    sy[3 * m + 1] = yk[((size_t)b * kM + p) * 3 + 1];
    sy[3 * m + 2] = yk[((size_t)b * kM + p) * 3 + 2];
  }
  if (threadIdx.x < 9) snum[threadIdx.x] = 0.f;
  if (threadIdx.x == 0) sden = 0.f;
  __syncthreads();
  float num[9] = {0, 0, 0, 0, 0, 0, 0, 0, 0};
  float den = 0.f;
  for (int n = threadIdx.x; n < kRN; n += blockDim.x) {
    int pn = p1[it * kRN + n];
    const unsigned char* wr = w + ((size_t)b * kN + pn) * kM;
    float t0 = 0.f, t1 = 0.f, t2 = 0.f; int c = 0;
    for (int m = 0; m < kRM; ++m) {
      if (wr[sp2[m]]) { t0 += sy[3 * m]; t1 += sy[3 * m + 1]; t2 += sy[3 * m + 2]; ++c; }
    }
    const float* xr = xk + ((size_t)b * kN + pn) * 3;
    float x0 = xr[0], x1 = xr[1], x2 = xr[2];
    num[0] += x0 * t0; num[1] += x0 * t1; num[2] += x0 * t2;
    num[3] += x1 * t0; num[4] += x1 * t1; num[5] += x1 * t2;
    num[6] += x2 * t0; num[7] += x2 * t1; num[8] += x2 * t2;
    den += (float)c;
  }
  for (int j = 0; j < 9; ++j) atomicAdd(&snum[j], num[j]);
  atomicAdd(&sden, den);
  __syncthreads();
  if (threadIdx.x < 9)
    models[(size_t)wg * 9 + threadIdx.x] = snum[threadIdx.x] / (sden + 1e-6f);
}

// ---------------------------------------------------------------- fused evaluator mean
// avoids materializing 256MB of errors; y tile + |y|^2 staged in LDS
__global__ __launch_bounds__(256) void avgerr_kernel(
    const float* __restrict__ xk, const float* __restrict__ yk,
    const unsigned char* __restrict__ w, const float* __restrict__ models,
    float* __restrict__ avg) {
  int wg = blockIdx.x;
  int b = wg / kIT;
  __shared__ float sy[kM * 3];
  __shared__ float syn[kM];
  __shared__ float ssum[256];
  __shared__ int scnt[256];
  for (int m = threadIdx.x; m < kM; m += blockDim.x) {
    float y0 = yk[((size_t)b * kM + m) * 3 + 0];
    float y1 = yk[((size_t)b * kM + m) * 3 + 1];
    float y2 = yk[((size_t)b * kM + m) * 3 + 2];
    sy[3 * m] = y0; sy[3 * m + 1] = y1; sy[3 * m + 2] = y2;
    syn[m] = y0 * y0 + y1 * y1 + y2 * y2;
  }
  __syncthreads();
  float md[9];
  for (int j = 0; j < 9; ++j) md[j] = models[(size_t)wg * 9 + j];
  float sum = 0.f; int cnt = 0;
  for (int n = threadIdx.x; n < kN; n += blockDim.x) {
    const float* xr = xk + ((size_t)b * kN + n) * 3;
    float x0 = xr[0], x1 = xr[1], x2 = xr[2];
    float p0 = x0 * md[0] + x1 * md[3] + x2 * md[6];
    float p1 = x0 * md[1] + x1 * md[4] + x2 * md[7];
    float p2 = x0 * md[2] + x1 * md[5] + x2 * md[8];
    float pn2 = p0 * p0 + p1 * p1 + p2 * p2;
    const unsigned char* wr = w + ((size_t)b * kN + n) * kM;
    for (int m = 0; m < kM; ++m) {
      if (wr[m]) {
        float d2 = pn2 + syn[m] -
                   2.f * (p0 * sy[3 * m] + p1 * sy[3 * m + 1] + p2 * sy[3 * m + 2]);
        sum += sqrtf(fmaxf(d2, 1e-12f));
        ++cnt;
      }
    }
  }
  ssum[threadIdx.x] = sum; scnt[threadIdx.x] = cnt;
  __syncthreads();
  for (int o = 128; o > 0; o >>= 1) {
    if (threadIdx.x < o) {
      ssum[threadIdx.x] += ssum[threadIdx.x + o];
      scnt[threadIdx.x] += scnt[threadIdx.x + o];
    }
    __syncthreads();
  }
  if (threadIdx.x == 0)
    avg[wg] = scnt[0] ? (ssum[0] / (float)scnt[0]) : dnan();
}

// ---------------------------------------------------------------- best model pick
__global__ void best_kernel(const float* __restrict__ avg,
                            const float* __restrict__ models,
                            int* __restrict__ bestit, float* __restrict__ outModel) {
  int b = threadIdx.x;
  if (b >= kB) return;
  int bi = 0; float bv = dinf();
  for (int i = 0; i < kIT; ++i) {
    float v = avg[b * kIT + i];
    if (v < bv) { bv = v; bi = i; }
  }
  bestit[b] = bi;
  for (int j = 0; j < 9; ++j)
    outModel[b * 9 + j] = models[(size_t)(b * kIT + bi) * 9 + j];
}

// ---------------------------------------------------------------- best errors (output)
__global__ void besterr_kernel(const float* __restrict__ xk,
                               const float* __restrict__ yk,
                               const unsigned char* __restrict__ w,
                               const float* __restrict__ outModel,
                               float* __restrict__ be) {
  int row = blockIdx.x;
  int b = row / kN;
  float md[9];
  for (int j = 0; j < 9; ++j) md[j] = outModel[b * 9 + j];
  const float* xr = xk + (size_t)row * 3;
  float x0 = xr[0], x1 = xr[1], x2 = xr[2];
  float p0 = x0 * md[0] + x1 * md[3] + x2 * md[6];
  float p1 = x0 * md[1] + x1 * md[4] + x2 * md[7];
  float p2 = x0 * md[2] + x1 * md[5] + x2 * md[8];
  float pn2 = p0 * p0 + p1 * p1 + p2 * p2;
  size_t base = (size_t)row * kM;
  for (int m = threadIdx.x; m < kM; m += blockDim.x) {
    const float* yr = yk + ((size_t)b * kM + m) * 3;
    float y0 = yr[0], y1 = yr[1], y2 = yr[2];
    float d2 = pn2 + y0 * y0 + y1 * y1 + y2 * y2 - 2.f * (p0 * y0 + p1 * y1 + p2 * y2);
    float e = sqrtf(fmaxf(d2, 1e-12f));
    be[base + m] = w[base + m] ? e : dnan();
  }
}

// ---------------------------------------------------------------- per-row 75% quantile
__global__ __launch_bounds__(256) void rowquant_kernel(
    const float* __restrict__ be, float* __restrict__ rowthr) {
  int row = blockIdx.x, tid = threadIdx.x;
  __shared__ float v[kM];
  __shared__ int ric[256];
  __shared__ float rfc[256];
  const float* r = be + (size_t)row * kM;
  int cnt = 0; float mx = 0.f;
  for (int m = tid; m < kM; m += 256) {
    float x = r[m];
    v[m] = x;
    if (x == x) { ++cnt; if (x > mx) mx = x; }
  }
  ric[tid] = cnt; rfc[tid] = mx;
  __syncthreads();
  for (int o = 128; o > 0; o >>= 1) {
    if (tid < o) { ric[tid] += ric[tid + o]; rfc[tid] = fmaxf(rfc[tid], rfc[tid + o]); }
    __syncthreads();
  }
  cnt = ric[0]; mx = rfc[0];
  __syncthreads();
  if (cnt == 0) { if (tid == 0) rowthr[row] = dnan(); return; }
  float h = kRansacThr * (float)(cnt - 1);
  int k = (int)h;
  float frac = h - (float)k;
  float lo = 0.f, hi = mx * 1.000001f + 1e-12f;
  for (int itb = 0; itb < 28; ++itb) {
    float mid = 0.5f * (lo + hi);
    int c = 0;
    for (int m = tid; m < kM; m += 256) c += (v[m] <= mid) ? 1 : 0;  // NaN -> false
    ric[tid] = c; __syncthreads();
    for (int o = 128; o > 0; o >>= 1) {
      if (tid < o) ric[tid] += ric[tid + o];
      __syncthreads();
    }
    c = ric[0]; __syncthreads();
    if (c >= k + 1) hi = mid; else lo = mid;
  }
  float vk = hi;
  int c = 0; float nxt = dinf();
  for (int m = tid; m < kM; m += 256) {
    float x = v[m];
    c += (x <= vk) ? 1 : 0;
    if (x > vk && x < nxt) nxt = x;
  }
  ric[tid] = c; rfc[tid] = nxt;
  __syncthreads();
  for (int o = 128; o > 0; o >>= 1) {
    if (tid < o) { ric[tid] += ric[tid + o]; rfc[tid] = fminf(rfc[tid], rfc[tid + o]); }
    __syncthreads();
  }
  if (tid == 0) {
    float vk1 = (k + 1 >= cnt) ? vk : ((ric[0] >= k + 2) ? vk : rfc[0]);
    rowthr[row] = vk + frac * (vk1 - vk);
  }
}

// ---------------------------------------------------------------- inliers (output)
__global__ void inlier_kernel(const float* __restrict__ be,
                              const float* __restrict__ rowthr,
                              float* __restrict__ outInl) {
  int row = blockIdx.x;
  float thr = rowthr[row];
  size_t base = (size_t)row * kM;
  for (int m = threadIdx.x; m < kM; m += blockDim.x) {
    float e = be[base + m];
    outInl[base + m] = (e < thr) ? 1.0f : 0.0f;  // NaN compares false
  }
}

// ================================================================ launch
extern "C" void kernel_launch(void* const* d_in, const int* in_sizes, int n_in,
                              void* d_out, int out_size, void* d_ws, size_t ws_size,
                              hipStream_t stream) {
  (void)in_sizes; (void)n_in; (void)out_size; (void)ws_size;
  const float* xk = (const float*)d_in[0];
  const float* xd = (const float*)d_in[1];
  const float* yk = (const float*)d_in[2];
  const float* yd = (const float*)d_in[3];
  const float* mask = (const float*)d_in[4];

  char* ws = (char*)d_ws;
  size_t off = 0;
  auto alloc = [&](size_t bytes) -> void* {
    void* p = ws + off;
    off = (off + bytes + 255) & ~(size_t)255;
    return p;
  };
  float* dist = (float*)alloc((size_t)kB * kN * kM * 4);
  float* xn = (float*)alloc((size_t)kB * kN * 4);
  float* yn = (float*)alloc((size_t)kB * kM * 4);
  float* thrq = (float*)alloc(kB * 4);
  int* ridx = (int*)alloc((size_t)kB * kN * 4);
  int* rrt = (int*)alloc((size_t)kB * kN * 4);
  int* cidx = (int*)alloc((size_t)kB * kM * 4);
  int* crt = (int*)alloc((size_t)kB * kM * 4);
  unsigned char* w = (unsigned char*)alloc((size_t)kB * kN * kM);
  int* p1 = (int*)alloc((size_t)kIT * kRN * 4);
  int* p2 = (int*)alloc((size_t)kIT * kRM * 4);
  float* models = (float*)alloc((size_t)kB * kIT * 9 * 4);
  float* avg = (float*)alloc((size_t)kB * kIT * 4);
  int* bestit = (int*)alloc(kB * 4);
  float* rowthr = (float*)alloc((size_t)kB * kN * 4);

  float* out = (float*)d_out;
  float* outInl = out;                                // (B,N,M) inliers as 0/1
  float* outModel = out + (size_t)kB * kN * kM;       // (B,3,3)
  float* outBE = outModel + (size_t)kB * 9;           // (B,N,M) best errors

  // 1) norms
  {
    int rows = kB * kN;
    norms_kernel<<<(rows * 32 + 255) / 256, 256, 0, stream>>>(xd, xn, rows);
    rows = kB * kM;
    norms_kernel<<<(rows * 32 + 255) / 256, 256, 0, stream>>>(yd, yn, rows);
  }
  // 2) WMMA GEMM + distance (async LDS staging)
  gemm_dist_kernel<<<dim3(kM / 64, kN / 32, kB), 256, 0, stream>>>(
      xd, yd, mask, xn, yn, dist);
  // 3) per-batch 1% quantile
  quant_kernel<<<kB, 1024, 0, stream>>>(dist, thrq);
  // 4) row / col top2 stats
  rowstat_kernel<<<(kB * kN * 32 + 255) / 256, 256, 0, stream>>>(dist, ridx, rrt);
  colstat_kernel<<<(kB * kM * 32 + 255) / 256, 256, 0, stream>>>(dist, cidx, crt);
  // 5) mutual-NN + ratio + threshold mask
  combine_kernel<<<kB * kN, 256, 0, stream>>>(dist, mask, ridx, rrt, cidx, crt,
                                              thrq, w);
  // 6) sampling permutations
  perm_kernel<<<1, 32, 0, stream>>>(p1, p2);
  // 7) per-(b,it) 3x3 solver
  solver_kernel<<<kB * kIT, 256, 0, stream>>>(xk, yk, w, p1, p2, models);
  // 8) fused evaluator nanmean (no 256MB errors tensor)
  avgerr_kernel<<<kB * kIT, 256, 0, stream>>>(xk, yk, w, models, avg);
  // 9) best model per batch (writes best_model output)
  best_kernel<<<1, 64, 0, stream>>>(avg, models, bestit, outModel);
  // 10) materialize best errors straight into d_out
  besterr_kernel<<<kB * kN, 256, 0, stream>>>(xk, yk, w, outModel, outBE);
  // 11) per-row 75% quantile
  rowquant_kernel<<<kB * kN, 256, 0, stream>>>(outBE, rowthr);
  // 12) inliers output
  inlier_kernel<<<kB * kN, 256, 0, stream>>>(outBE, rowthr, outInl);
}
